// LMLT_40381282517343
// MI455X (gfx1250) — matrix operand order repaired and checked by reading, not compiled, not run on hardware
//
#include <hip/hip_runtime.h>
#include <hip/hip_bf16.h>

typedef __attribute__((ext_vector_type(16))) _Float16 v16h;
typedef __attribute__((ext_vector_type(8)))  float    v8f;

#define IMG_H  256
#define IMG_HW 65536
#define NB     8   // batch

__device__ __forceinline__ float gelu_exact(float x) {
    return 0.5f * x * (1.0f + erff(x * 0.70710678118654752f));
}
__device__ __forceinline__ float elu1(float x) {   // elu(x)+1
    return x > 0.0f ? x + 1.0f : expf(x);
}

// ---------------------------------------------------------------------------
// to_feat: 3x3 conv 3->12 pad1, writes feat and y (residual copy)
// ---------------------------------------------------------------------------
__global__ void k_to_feat(const float* __restrict__ x, const float* __restrict__ W,
                          const float* __restrict__ bb, float* __restrict__ feat,
                          float* __restrict__ y) {
    int idx = blockIdx.x * blockDim.x + threadIdx.x;       // over NB*IMG_HW
    int b = idx / IMG_HW, p = idx - b * IMG_HW;
    int ph = p >> 8, pw = p & 255;
    float a[12];
    #pragma unroll
    for (int co = 0; co < 12; ++co) a[co] = bb[co];
    #pragma unroll
    for (int ti = -1; ti <= 1; ++ti)
        #pragma unroll
        for (int tj = -1; tj <= 1; ++tj) {
            int hh = ph + ti, ww = pw + tj;
            if (hh < 0 || hh >= IMG_H || ww < 0 || ww >= IMG_H) continue;
            int wi = (ti + 1) * 3 + (tj + 1);
            #pragma unroll
            for (int ci = 0; ci < 3; ++ci) {
                float xv = x[((size_t)b * 3 + ci) * IMG_HW + hh * IMG_H + ww];
                #pragma unroll
                for (int co = 0; co < 12; ++co)
                    a[co] += W[(co * 3 + ci) * 9 + wi] * xv;
            }
        }
    #pragma unroll
    for (int co = 0; co < 12; ++co) {
        size_t oi = ((size_t)b * 12 + co) * IMG_HW + p;
        feat[oi] = a[co];
        y[oi] = a[co];
    }
}

// ---------------------------------------------------------------------------
// channel layer-norm over C=12 per pixel
// ---------------------------------------------------------------------------
__global__ void k_layernorm(const float* __restrict__ x, const float* __restrict__ w,
                            const float* __restrict__ bb, float* __restrict__ o) {
    int idx = blockIdx.x * blockDim.x + threadIdx.x;
    int b = idx / IMG_HW, p = idx - b * IMG_HW;
    const float* xb = x + (size_t)b * 12 * IMG_HW + p;
    float v[12], mu = 0.0f;
    #pragma unroll
    for (int c = 0; c < 12; ++c) { v[c] = xb[(size_t)c * IMG_HW]; mu += v[c]; }
    mu *= (1.0f / 12.0f);
    float var = 0.0f;
    #pragma unroll
    for (int c = 0; c < 12; ++c) { float d = v[c] - mu; var += d * d; }
    var *= (1.0f / 12.0f);
    float inv = rsqrtf(var + 1e-6f);
    float* ob = o + (size_t)b * 12 * IMG_HW + p;
    #pragma unroll
    for (int c = 0; c < 12; ++c) ob[(size_t)c * IMG_HW] = w[c] * (v[c] - mu) * inv + bb[c];
}

// ---------------------------------------------------------------------------
// max-pool sxs (stride s) of channels [coff, coff+d) of a 12-ch full-res tensor
// ---------------------------------------------------------------------------
__global__ void k_maxpool(const float* __restrict__ x, int coff, int d, int s, int Hp,
                          float* __restrict__ o) {
    int idx = blockIdx.x * blockDim.x + threadIdx.x;       // NB*Hp*Hp
    int HWp = Hp * Hp;
    int b = idx / HWp, pp = idx - b * HWp;
    int ph = pp / Hp, pw = pp - ph * Hp;
    for (int c = 0; c < d; ++c) {
        const float* src = x + ((size_t)b * 12 + coff + c) * IMG_HW + (ph * s) * IMG_H + pw * s;
        float m = -3.402823466e38f;
        for (int i = 0; i < s; ++i)
            for (int j = 0; j < s; ++j)
                m = fmaxf(m, src[i * IMG_H + j]);
        o[((size_t)b * d + c) * HWp + pp] = m;
    }
}

// ---------------------------------------------------------------------------
// Attention pass 1 (templated on D): per-pixel recompute k_e = elu1(k+dw3x3(k)),
// v_p = v + dw3x3(v); accumulate sum_k[e], sum_kv[e][f] via deterministic
// wave32 shuffle tree + fixed-order block combine (no atomics).
// ---------------------------------------------------------------------------
template <int D>
__global__ void k_attn_reduce(const float* __restrict__ xi, int cinStride, int Hp,
                              const float* __restrict__ qw, const float* __restrict__ qb,
                              const float* __restrict__ dwk, const float* __restrict__ dwkb,
                              const float* __restrict__ dwv, const float* __restrict__ dwvb,
                              float* __restrict__ part) {
    __shared__ float sp[8 * 42];
    int tid = threadIdx.x;
    int idx = blockIdx.x * 256 + tid;
    int HWp = Hp * Hp;
    int b = idx / HWp, p = idx - b * HWp;
    int ph = p / Hp, pw = p - ph * Hp;

    float kc[D], vc[D], kd[D], vd[D];
    #pragma unroll
    for (int e = 0; e < D; ++e) { kc[e] = vc[e] = kd[e] = vd[e] = 0.0f; }
    #pragma unroll
    for (int ti = -1; ti <= 1; ++ti)
        #pragma unroll
        for (int tj = -1; tj <= 1; ++tj) {
            int hh = ph + ti, ww = pw + tj;
            if (hh < 0 || hh >= Hp || ww < 0 || ww >= Hp) continue;
            float xt[D];
            #pragma unroll
            for (int c = 0; c < D; ++c)
                xt[c] = xi[((size_t)b * cinStride + c) * HWp + hh * Hp + ww];
            int wi = (ti + 1) * 3 + (tj + 1);
            #pragma unroll
            for (int e = 0; e < D; ++e) {
                float kt = qb[D + e], vt = qb[2 * D + e];
                const float* wk = qw + (size_t)(D + e) * D;
                const float* wv = qw + (size_t)(2 * D + e) * D;
                #pragma unroll
                for (int c = 0; c < D; ++c) { kt += wk[c] * xt[c]; vt += wv[c] * xt[c]; }
                kd[e] += dwk[e * 9 + wi] * kt;
                vd[e] += dwv[e * 9 + wi] * vt;
                if (ti == 0 && tj == 0) { kc[e] = kt; vc[e] = vt; }
            }
        }
    float ke[D], vp[D];
    #pragma unroll
    for (int e = 0; e < D; ++e) {
        ke[e] = elu1(kc[e] + kd[e] + dwkb[e]);
        vp[e] = vc[e] + vd[e] + dwvb[e];
    }
    int lane = tid & 31, wv32 = tid >> 5;
    #pragma unroll
    for (int j = 0; j < 42; ++j) {
        bool live;
        float val = 0.0f;
        if (j < 6) { live = (j < D); if (live) val = ke[j < D ? j : 0]; }
        else {
            int e = (j - 6) / 6, f = (j - 6) % 6;
            live = (e < D && f < D);
            if (live) val = ke[e < D ? e : 0] * vp[f < D ? f : 0];
        }
        if (live) {
            #pragma unroll
            for (int off = 16; off; off >>= 1) val += __shfl_down(val, off, 32);
        }
        if (lane == 0) sp[wv32 * 42 + j] = val;
    }
    __syncthreads();
    if (tid < 42) {
        float s = 0.0f;
        #pragma unroll
        for (int w = 0; w < 8; ++w) s += sp[w * 42 + tid];
        part[(size_t)blockIdx.x * 64 + tid] = s;
    }
}

// fixed-order finalize: acc[b*64 + j] = sum over this image's block partials
__global__ void k_acc_finalize(const float* __restrict__ part, int bpi, float* __restrict__ acc) {
    int b = blockIdx.x, j = threadIdx.x;       // <<<8, 64>>>
    if (j < 42) {
        float s = 0.0f;
        for (int k = 0; k < bpi; ++k) s += part[((size_t)(b * bpi + k)) * 64 + j];
        acc[(size_t)b * 64 + j] = s;
    }
}

// ---------------------------------------------------------------------------
// Attention pass 2 (templated on D): recompute q_e, v_p, lepe (5x5 of v_pre),
// o; combine with km/kv stats; proj 1x1; write branch output at pooled res.
// ---------------------------------------------------------------------------
template <int D>
__global__ void k_attn_combine(const float* __restrict__ xi, int cinStride, int Hp,
                               const float* __restrict__ qw, const float* __restrict__ qb,
                               const float* __restrict__ dwq, const float* __restrict__ dwqb,
                               const float* __restrict__ dwv, const float* __restrict__ dwvb,
                               const float* __restrict__ lw, const float* __restrict__ lb,
                               const float* __restrict__ prw, const float* __restrict__ prb,
                               const float* __restrict__ acc, float* __restrict__ brout) {
    int idx = blockIdx.x * 256 + threadIdx.x;
    int HWp = Hp * Hp;
    int b = idx / HWp, p = idx - b * HWp;
    int ph = p / Hp, pw = p - ph * Hp;
    float fN = (float)HWp;
    float scale = rsqrtf((float)D);

    const float* a = acc + (size_t)b * 64;
    float km[D], kv[D * D];
    #pragma unroll
    for (int e = 0; e < D; ++e) km[e] = a[e] / fN;
    float kvs = scale / fN;
    #pragma unroll
    for (int e = 0; e < D; ++e)
        #pragma unroll
        for (int f = 0; f < D; ++f) kv[e * D + f] = a[6 + e * 6 + f] * kvs;

    float qc[D], vc[D], oo[D], qd[D], vd[D], lep[D];
    #pragma unroll
    for (int e = 0; e < D; ++e) { qc[e] = vc[e] = oo[e] = qd[e] = vd[e] = lep[e] = 0.0f; }

    #pragma unroll
    for (int ti = -2; ti <= 2; ++ti)
        #pragma unroll
        for (int tj = -2; tj <= 2; ++tj) {
            int hh = ph + ti, ww = pw + tj;
            if (hh < 0 || hh >= Hp || ww < 0 || ww >= Hp) continue;
            float xt[D];
            #pragma unroll
            for (int c = 0; c < D; ++c)
                xt[c] = xi[((size_t)b * cinStride + c) * HWp + hh * Hp + ww];
            int li = (ti + 2) * 5 + (tj + 2);
            constexpr int T3LO = -1, T3HI = 1;
            bool in3 = (ti >= T3LO && ti <= T3HI && tj >= T3LO && tj <= T3HI);
            int w3 = (ti + 1) * 3 + (tj + 1);
            #pragma unroll
            for (int e = 0; e < D; ++e) {
                float vt = qb[2 * D + e];
                const float* wv = qw + (size_t)(2 * D + e) * D;
                #pragma unroll
                for (int c = 0; c < D; ++c) vt += wv[c] * xt[c];
                lep[e] += lw[e * 25 + li] * vt;
                if (in3) {
                    float qt = qb[e];
                    const float* wq = qw + (size_t)e * D;
                    #pragma unroll
                    for (int c = 0; c < D; ++c) qt += wq[c] * xt[c];
                    qd[e] += dwq[e * 9 + w3] * qt;
                    vd[e] += dwv[e * 9 + w3] * vt;
                    if (ti == 0 && tj == 0) {
                        qc[e] = qt; vc[e] = vt;
                        float ot = qb[3 * D + e];
                        const float* wo = qw + (size_t)(3 * D + e) * D;
                        #pragma unroll
                        for (int c = 0; c < D; ++c) ot += wo[c] * xt[c];
                        oo[e] = ot;
                    }
                }
            }
        }
    float qe[D], vpv[D], z = 0.0f, vmean = 0.0f;
    #pragma unroll
    for (int e = 0; e < D; ++e) {
        qe[e] = elu1(qc[e] + qd[e] + dwqb[e]);
        vpv[e] = vc[e] + vd[e] + dwvb[e];
        z += qe[e] * km[e];
        vmean += vpv[e];
    }
    z *= scale;
    vmean *= (1.0f / (float)D);
    float gmul = 1.0f + 1.0f / (z + 1e-6f);
    float t[D];
    #pragma unroll
    for (int e = 0; e < D; ++e) {
        float r = 0.0f;
        #pragma unroll
        for (int f = 0; f < D; ++f) r += qe[f] * kv[f * D + e];
        t[e] = (r * gmul - z * vmean + (lep[e] + lb[e])) * oo[e];
    }
    #pragma unroll
    for (int e = 0; e < D; ++e) {
        float r = prb[e];
        #pragma unroll
        for (int f = 0; f < D; ++f) r += prw[e * D + f] * t[f];
        brout[((size_t)b * D + e) * HWp + p] = r;
    }
}

// ---------------------------------------------------------------------------
// Fused gate + nearest-upsample gather + aggr 1x1 (WMMA) + residual add.
// One wave: 16 pixels x 12 out channels. A and B both use contiguous k = 16*laneHi + h.
// ---------------------------------------------------------------------------
__global__ void k_aggr_wmma(const float* __restrict__ tn,
                            const float* __restrict__ br0, const float* __restrict__ br1,
                            const float* __restrict__ br2, const float* __restrict__ br3,
                            const float* __restrict__ gw, const float* __restrict__ gb,
                            const float* __restrict__ aw, const float* __restrict__ ab,
                            float* __restrict__ y) {
    int lane = threadIdx.x & 31;
    int wid = (blockIdx.x * blockDim.x + threadIdx.x) >> 5;
    int g = wid << 4;
    int b = g / IMG_HW, p = g - b * IMG_HW;
    int n = lane & 15, laneHi = lane >> 4;

    v16h Am, Bm;
    #pragma unroll
    for (int h = 0; h < 16; ++h) {
        int k = (laneHi << 4) + h;
        Am[h] = (_Float16)((n < 12 && k < 12) ? aw[n * 12 + k] : 0.0f);
        Bm[h] = (_Float16)0.0f;
    }
    if (laneHi == 0) {
        int pp = p + n, ph = pp >> 8, pw = pp & 255;
        float xn[12];
        #pragma unroll
        for (int c = 0; c < 12; ++c) xn[c] = tn[((size_t)b * 12 + c) * IMG_HW + pp];
        float val[5];
        #pragma unroll
        for (int o6 = 0; o6 < 5; ++o6) {
            float s = gb[o6];
            #pragma unroll
            for (int c = 0; c < 12; ++c) s += gw[o6 * 12 + c] * xn[c];
            val[o6] = s;
        }
        float inb[12];
        { int bp = (ph >> 3) * 32 + (pw >> 3); inb[0] = br0[(size_t)b * 1024 + bp] * val[0]; }
        { int bp = (ph >> 2) * 64 + (pw >> 2);
          #pragma unroll
          for (int c = 0; c < 2; ++c) inb[1 + c] = br1[((size_t)b * 2 + c) * 4096 + bp] * val[1 + c]; }
        { int bp = (ph >> 1) * 128 + (pw >> 1);
          #pragma unroll
          for (int c = 0; c < 3; ++c) inb[3 + c] = br2[((size_t)b * 3 + c) * 16384 + bp] * val[2 + c]; }
        #pragma unroll
        for (int c = 0; c < 6; ++c) inb[6 + c] = br3[((size_t)b * 6 + c) * IMG_HW + pp];
        #pragma unroll
        for (int h = 0; h < 12; ++h) Bm[h] = (_Float16)inb[h];
    }
    v8f C;
    #pragma unroll
    for (int i = 0; i < 8; ++i) { int mm = i + (laneHi << 3); C[i] = (mm < 12) ? ab[mm] : 0.0f; }
    C = __builtin_amdgcn_wmma_f32_16x16x32_f16(false, Am, false, Bm, (short)0, C, false, false);
    #pragma unroll
    for (int i = 0; i < 8; ++i) {
        int mm = i + (laneHi << 3);
        if (mm < 12) {
            size_t oi = ((size_t)b * 12 + mm) * IMG_HW + p + n;
            y[oi] += C[i];
        }
    }
}

// ---------------------------------------------------------------------------
// Generic 1x1 conv via WMMA, compile-time shapes, MT pixel-tiles per wave
// (weight pack amortized over MT back-to-back WMMAs). Optional 2nd input
// covers channels [CINA, CIN) (pcfn concat). EPI: 0=none, 1=exact gelu.
// ---------------------------------------------------------------------------
template <int CIN, int CINA, int COUT, int EPI, bool RESID, int MT>
__global__ void k_conv1x1_wmma(const float* __restrict__ xA, int strideA,
                               const float* __restrict__ xB, int strideB,
                               const float* __restrict__ W, const float* __restrict__ bias,
                               float* __restrict__ outp, int strideO,
                               const float* __restrict__ resid, int strideR,
                               int HW_) {
    int lane = threadIdx.x & 31;
    int wid = (blockIdx.x * blockDim.x + threadIdx.x) >> 5;
    int g = wid * (16 * MT);
    int b = g / HW_, p0 = g - b * HW_;
    int n = lane & 15, laneHi = lane >> 4;

    v16h Am;
    #pragma unroll
    for (int h = 0; h < 16; ++h) {
        int k = (laneHi << 4) + h;
        Am[h] = (_Float16)((n < COUT && k < CIN) ? W[n * CIN + k] : 0.0f);
    }
    #pragma unroll
    for (int t = 0; t < MT; ++t) {
        int p = p0 + 16 * t;
        v16h Bm;
        #pragma unroll
        for (int h = 0; h < 16; ++h) {
            int k = (laneHi << 4) + h;
            float xv = 0.0f;
            if (k < CIN) {
                if (k < CINA) xv = xA[((size_t)b * strideA + k) * HW_ + p + n];
                else          xv = xB[((size_t)b * strideB + (k - CINA)) * HW_ + p + n];
            }
            Bm[h] = (_Float16)xv;
        }
        v8f C;
        #pragma unroll
        for (int i = 0; i < 8; ++i) {
            int mm = i + (laneHi << 3);
            C[i] = (mm < COUT) ? bias[mm] : 0.0f;
        }
        C = __builtin_amdgcn_wmma_f32_16x16x32_f16(false, Am, false, Bm, (short)0, C, false, false);
        #pragma unroll
        for (int i = 0; i < 8; ++i) {
            int mm = i + (laneHi << 3);
            if (mm < COUT) {
                float r = C[i];
                if (EPI == 1) r = gelu_exact(r);
                if (RESID) r += resid[((size_t)b * strideR + mm) * HW_ + p + n];
                outp[((size_t)b * strideO + mm) * HW_ + p + n] = r;
            }
        }
    }
}

// ---------------------------------------------------------------------------
// pcfn c1: 3x3 conv 6->6 pad1 on t_hid channels 0..5 (stride 24), exact gelu
// ---------------------------------------------------------------------------
__global__ void k_c1(const float* __restrict__ x, const float* __restrict__ W,
                     const float* __restrict__ bb, float* __restrict__ o) {
    int idx = blockIdx.x * blockDim.x + threadIdx.x;
    int b = idx / IMG_HW, p = idx - b * IMG_HW;
    int ph = p >> 8, pw = p & 255;
    float a[6];
    #pragma unroll
    for (int co = 0; co < 6; ++co) a[co] = bb[co];
    #pragma unroll
    for (int ti = -1; ti <= 1; ++ti)
        #pragma unroll
        for (int tj = -1; tj <= 1; ++tj) {
            int hh = ph + ti, ww = pw + tj;
            if (hh < 0 || hh >= IMG_H || ww < 0 || ww >= IMG_H) continue;
            int wi = (ti + 1) * 3 + (tj + 1);
            #pragma unroll
            for (int ci = 0; ci < 6; ++ci) {
                float xv = x[((size_t)b * 24 + ci) * IMG_HW + hh * IMG_H + ww];
                #pragma unroll
                for (int co = 0; co < 6; ++co)
                    a[co] += W[(co * 6 + ci) * 9 + wi] * xv;
            }
        }
    #pragma unroll
    for (int co = 0; co < 6; ++co)
        o[((size_t)b * 6 + co) * IMG_HW + p] = gelu_exact(a[co]);
}

// ---------------------------------------------------------------------------
// to_img: 3x3 conv 12->12 pad1 on (y+feat), fused pixel-shuffle (r=2) store
// ---------------------------------------------------------------------------
__global__ void k_to_img(const float* __restrict__ y, const float* __restrict__ feat,
                         const float* __restrict__ W, const float* __restrict__ bb,
                         float* __restrict__ out) {
    int idx = blockIdx.x * blockDim.x + threadIdx.x;
    int b = idx / IMG_HW, p = idx - b * IMG_HW;
    int ph = p >> 8, pw = p & 255;
    float a[12];
    #pragma unroll
    for (int co = 0; co < 12; ++co) a[co] = bb[co];
    #pragma unroll
    for (int ti = -1; ti <= 1; ++ti)
        #pragma unroll
        for (int tj = -1; tj <= 1; ++tj) {
            int hh = ph + ti, ww = pw + tj;
            if (hh < 0 || hh >= IMG_H || ww < 0 || ww >= IMG_H) continue;
            int wi = (ti + 1) * 3 + (tj + 1);
            #pragma unroll
            for (int ci = 0; ci < 12; ++ci) {
                size_t si = ((size_t)b * 12 + ci) * IMG_HW + hh * IMG_H + ww;
                float xv = y[si] + feat[si];
                #pragma unroll
                for (int co = 0; co < 12; ++co)
                    a[co] += W[(co * 12 + ci) * 9 + wi] * xv;
            }
        }
    #pragma unroll
    for (int co = 0; co < 12; ++co) {
        int cc = co >> 2, r0 = (co >> 1) & 1, r1 = co & 1;
        out[(((size_t)b * 3 + cc) * 512 + (ph * 2 + r0)) * 512 + (pw * 2 + r1)] = a[co];
    }
}

// ---------------------------------------------------------------------------
extern "C" void kernel_launch(void* const* d_in, const int* in_sizes, int n_in,
                              void* d_out, int out_size, void* d_ws, size_t ws_size,
                              hipStream_t stream) {
    (void)in_sizes; (void)n_in; (void)out_size; (void)ws_size;
    auto in = [&](int i) { return (const float*)d_in[i]; };
    float* ws = (float*)d_ws;

    const size_t NFEAT = (size_t)NB * 12 * IMG_HW;          // 6,291,456
    const size_t NHID  = (size_t)NB * 24 * IMG_HW;          // 12,582,912
    float* feat = ws;
    float* y    = feat + NFEAT;
    float* tno  = y + NFEAT;            // norm buffer; also t_p (6ch) region in pcfn
    float* thid = tno + NFEAT;          // 24ch hidden; gbc scratch aliases inside
    float* xi0 = thid;                                  // 8*1*1024
    float* xi1 = xi0 + (size_t)NB * 1024;               // 8*2*4096
    float* xi2 = xi1 + (size_t)NB * 2 * 4096;           // 8*3*16384
    float* br0 = xi2 + (size_t)NB * 3 * 16384;          // 8*1*1024
    float* br1 = br0 + (size_t)NB * 1024;               // 8*2*4096
    float* br2 = br1 + (size_t)NB * 2 * 4096;           // 8*3*16384
    float* br3 = br2 + (size_t)NB * 3 * 16384;          // 8*6*65536 (fits in NHID)
    float* acc  = thid + NHID;                          // 4*8*64
    float* part = acc + 4 * 8 * 64;                     // up to 2048*64

    const int PIX_BLOCKS = (NB * IMG_HW) / 256;              // 2048
    const int AGGR_BLOCKS = (NB * IMG_HW / 16) * 32 / 256;   // 4096 (1 tile/wave)
    const int CONV_BLOCKS = (NB * IMG_HW / 64) * 32 / 256;   // 1024 (4 tiles/wave)

    k_to_feat<<<PIX_BLOCKS, 256, 0, stream>>>(in(0), in(1), in(2), feat, y);

    for (int blk = 0; blk < NB; ++blk) {
        int base = 3 + 62 * blk;
        const float *n1w = in(base + 0), *n1b = in(base + 1);
        const float *n2w = in(base + 2), *n2b = in(base + 3);
        const float *gw = in(base + 4), *gb = in(base + 5);
        const float *aw = in(base + 6), *ab = in(base + 7);
        const float *c0w = in(base + 56), *c0b = in(base + 57);
        const float *c1w = in(base + 58), *c1b = in(base + 59);
        const float *c2w = in(base + 60), *c2b = in(base + 61);

        // ---- gbc ----
        k_layernorm<<<PIX_BLOCKS, 256, 0, stream>>>(y, n1w, n1b, tno);
        k_maxpool<<<(NB * 1024) / 256, 256, 0, stream>>>(tno, 0, 1, 8, 32, xi0);
        k_maxpool<<<(NB * 4096) / 256, 256, 0, stream>>>(tno, 1, 2, 4, 64, xi1);
        k_maxpool<<<(NB * 16384) / 256, 256, 0, stream>>>(tno, 3, 3, 2, 128, xi2);

        struct BI { const float* xi; int cs; int Hp; float* br; } bi[4] = {
            { xi0, 1, 32, br0 }, { xi1, 2, 64, br1 },
            { xi2, 3, 128, br2 }, { tno + (size_t)6 * IMG_HW, 12, 256, br3 }
        };
        for (int i = 0; i < 4; ++i) {
            int ab2 = base + 8 + 12 * i;
            const float *qw = in(ab2 + 0), *qb = in(ab2 + 1);
            const float *lw = in(ab2 + 2), *lb = in(ab2 + 3);
            const float *prw = in(ab2 + 4), *prb = in(ab2 + 5);
            const float *dwq = in(ab2 + 6), *dwqb = in(ab2 + 7);
            const float *dwk = in(ab2 + 8), *dwkb = in(ab2 + 9);
            const float *dwv = in(ab2 + 10), *dwvb = in(ab2 + 11);
            int HWp = bi[i].Hp * bi[i].Hp;
            int nblk = (NB * HWp) / 256;
            int bpi = HWp / 256;
            float* accB = acc + (size_t)i * NB * 64;
            switch (i) {
            case 0:
                k_attn_reduce<1><<<nblk, 256, 0, stream>>>(bi[i].xi, bi[i].cs, bi[i].Hp,
                    qw, qb, dwk, dwkb, dwv, dwvb, part);
                k_acc_finalize<<<NB, 64, 0, stream>>>(part, bpi, accB);
                k_attn_combine<1><<<nblk, 256, 0, stream>>>(bi[i].xi, bi[i].cs, bi[i].Hp,
                    qw, qb, dwq, dwqb, dwv, dwvb, lw, lb, prw, prb, accB, bi[i].br);
                break;
            case 1:
                k_attn_reduce<2><<<nblk, 256, 0, stream>>>(bi[i].xi, bi[i].cs, bi[i].Hp,
                    qw, qb, dwk, dwkb, dwv, dwvb, part);
                k_acc_finalize<<<NB, 64, 0, stream>>>(part, bpi, accB);
                k_attn_combine<2><<<nblk, 256, 0, stream>>>(bi[i].xi, bi[i].cs, bi[i].Hp,
                    qw, qb, dwq, dwqb, dwv, dwvb, lw, lb, prw, prb, accB, bi[i].br);
                break;
            case 2:
                k_attn_reduce<3><<<nblk, 256, 0, stream>>>(bi[i].xi, bi[i].cs, bi[i].Hp,
                    qw, qb, dwk, dwkb, dwv, dwvb, part);
                k_acc_finalize<<<NB, 64, 0, stream>>>(part, bpi, accB);
                k_attn_combine<3><<<nblk, 256, 0, stream>>>(bi[i].xi, bi[i].cs, bi[i].Hp,
                    qw, qb, dwq, dwqb, dwv, dwvb, lw, lb, prw, prb, accB, bi[i].br);
                break;
            default:
                k_attn_reduce<6><<<nblk, 256, 0, stream>>>(bi[i].xi, bi[i].cs, bi[i].Hp,
                    qw, qb, dwk, dwkb, dwv, dwvb, part);
                k_acc_finalize<<<NB, 64, 0, stream>>>(part, bpi, accB);
                k_attn_combine<6><<<nblk, 256, 0, stream>>>(bi[i].xi, bi[i].cs, bi[i].Hp,
                    qw, qb, dwq, dwqb, dwv, dwvb, lw, lb, prw, prb, accB, bi[i].br);
                break;
            }
        }
        k_aggr_wmma<<<AGGR_BLOCKS, 256, 0, stream>>>(tno, br0, br1, br2, br3,
                                                     gw, gb, aw, ab, y);

        // ---- pcfn ----
        k_layernorm<<<PIX_BLOCKS, 256, 0, stream>>>(y, n2w, n2b, tno);
        // c0: 12->24 + gelu (two M tiles: 16 + 8)
        k_conv1x1_wmma<12, 12, 16, 1, false, 4><<<CONV_BLOCKS, 256, 0, stream>>>(
            tno, 12, nullptr, 0, c0w, c0b, thid, 24, nullptr, 0, IMG_HW);
        k_conv1x1_wmma<12, 12, 8, 1, false, 4><<<CONV_BLOCKS, 256, 0, stream>>>(
            tno, 12, nullptr, 0, c0w + 16 * 12, c0b + 16,
            thid + (size_t)16 * IMG_HW, 24, nullptr, 0, IMG_HW);
        // c1: 3x3 6->6 + gelu -> t_p (aliases tno; norm2 already consumed)
        k_c1<<<PIX_BLOCKS, 256, 0, stream>>>(thid, c1w, c1b, tno);
        // c2: concat(t_p[6], thid[6:24]) -> 12, + residual y
        k_conv1x1_wmma<24, 6, 12, 0, true, 4><<<CONV_BLOCKS, 256, 0, stream>>>(
            tno, 6, thid + (size_t)6 * IMG_HW, 24, c2w, c2b,
            y, 12, y, 12, IMG_HW);
    }

    k_to_img<<<PIX_BLOCKS, 256, 0, stream>>>(y, feat, in(499), in(500), (float*)d_out);
}
